// ASPPConv_65335042507524
// MI455X (gfx1250) — matrix operand order repaired and checked by reading, not compiled, not run on hardware
//
#include <hip/hip_runtime.h>
#include <hip/hip_bf16.h>
#include <stdint.h>

#define K_TAPS   9
#define DILATION 6
#define PADH     24          // DIL * (K/2)
#define B_SZ     8
#define CIN      256
#define COUT     256
#define NPTS     8192
#define BN_EPS   1e-5f

#define MTILE 128
#define NTILE 128
#define KSLC  32
#define XW    176            // NTILE + 2*PADH
#define XHPAD 19             // half2 row stride: odd -> conflict-free b32 frag reads
#define APAD  18             // u32 row stride: even -> aligned + conflict-free b64 reads

typedef __attribute__((ext_vector_type(16))) _Float16 v16h;
typedef __attribute__((ext_vector_type(2)))  _Float16 half2_t;
typedef __attribute__((ext_vector_type(8)))  float    f32x8;
typedef int v2i __attribute__((vector_size(8)));       // b64 payload type for async LDS

#define ASG __attribute__((address_space(1)))
#define ASL __attribute__((address_space(3)))

// ---- optional gfx1250 async-to-LDS path for W staging (raw byte copy) ------
#ifndef USE_ASYNC_A
#if defined(__has_builtin)
#if __has_builtin(__builtin_amdgcn_global_load_async_to_lds_b64) && \
    __has_builtin(__builtin_amdgcn_s_wait_asynccnt)
#define USE_ASYNC_A 1
#endif
#endif
#endif
#ifndef USE_ASYNC_A
#define USE_ASYNC_A 0
#endif

#if USE_ASYNC_A
#define ASYNC_WAIT() __builtin_amdgcn_s_wait_asynccnt(0)
#else
#define ASYNC_WAIT()
#endif

// ---------------------------------------------------------------------------
// Kernel 1: data-dependent tap weights g[b][k][n] (zero for out-of-range taps)
// ---------------------------------------------------------------------------
__global__ __launch_bounds__(256) void tap_weight_kernel(
    const float* __restrict__ coords, const float* __restrict__ rot,
    const float* __restrict__ dist, float* __restrict__ g) {
  int idx = blockIdx.x * 256 + threadIdx.x;          // over B*N
  if (idx >= B_SZ * NPTS) return;
  int b = idx / NPTS, n = idx % NPTS;
  const float* cb = coords + (size_t)b * 3 * NPTS;
  const float* rb = rot    + (size_t)b * 3 * NPTS;
  const float* db = dist   + (size_t)b * NPTS;
  float c0x = cb[n], c0y = cb[NPTS + n], c0z = cb[2 * NPTS + n];
  float r0x = rb[n], r0y = rb[NPTS + n], r0z = rb[2 * NPTS + n];
  float d0  = db[n];
  float r0n = r0x * r0x + r0y * r0y + r0z * r0z;
#pragma unroll
  for (int k = 0; k < K_TAPS; ++k) {
    int j = n + k * DILATION - PADH;
    float gv = 0.f;
    if (j >= 0 && j < NPTS) {
      float dx = c0x - cb[j], dy = c0y - cb[NPTS + j], dz = c0z - cb[2 * NPTS + j];
      float dc = dx * dx + dy * dy + dz * dz;
      float dd = (d0 - db[j]) * (d0 - db[j]);
      float ge = expf(-(dc + dd) * 0.5f);            // sigma = 1
      float rjx = rb[j], rjy = rb[NPTS + j], rjz = rb[2 * NPTS + j];
      float num = r0x * rjx + r0y * rjy + r0z * rjz;
      float den = sqrtf(r0n * (rjx * rjx + rjy * rjy + rjz * rjz)) + 1e-8f;
      gv = ge * fabsf(num / den);
    }
    g[((size_t)b * K_TAPS + k) * NPTS + n] = gv;
  }
}

// ---------------------------------------------------------------------------
// Kernel 2: pack W[o][c][k] (f32) -> Wh[k][o][c] (f16, c contiguous)
// ---------------------------------------------------------------------------
__global__ __launch_bounds__(256) void pack_w_kernel(
    const float* __restrict__ W, uint16_t* __restrict__ Wh) {
  int idx = blockIdx.x * 256 + threadIdx.x;
  if (idx >= K_TAPS * COUT * CIN) return;
  int k = idx / (COUT * CIN);
  int rem = idx % (COUT * CIN);
  int o = rem / CIN, c = rem % CIN;
  union { _Float16 h; uint16_t u; } cv;
  cv.h = (_Float16)W[((size_t)o * CIN + c) * K_TAPS + k];   // RNE
  Wh[idx] = cv.u;
}

// ---------------------------------------------------------------------------
// W-slice stager: one tap's A tile (128 x 32 f16 as u32 pairs) into LDS
// ---------------------------------------------------------------------------
__device__ __forceinline__ void stage_A(half2_t (*dst)[APAD],
                                        const uint32_t* __restrict__ Wh32,
                                        int k, int o0, int c0, int tid) {
#if USE_ASYNC_A
  for (int idx = tid; idx < MTILE * 8; idx += 256) {
    int m = idx >> 3, q = idx & 7;                    // q: 8-byte chunk in row
    size_t g32 = ((size_t)k * COUT + o0 + m) * (CIN / 2) + (c0 >> 1) + 2 * q;
    __builtin_amdgcn_global_load_async_to_lds_b64(
        (ASG v2i*)(Wh32 + g32), (ASL v2i*)&dst[m][2 * q], 0, 0);
  }
#else
  for (int idx = tid; idx < MTILE * 16; idx += 256) {
    int m = idx >> 4, p = idx & 15;
    uint32_t w = Wh32[((size_t)k * COUT + o0 + m) * (CIN / 2) + (c0 >> 1) + p];
    ((uint32_t*)&dst[m][0])[p] = w;
  }
#endif
}

// ---------------------------------------------------------------------------
// Kernel 3: fused tap-weighted dilated conv as one GEMM over (c,k), f16 WMMA.
//   C[128x128] per WG; 8 waves each own 64x32 via v_wmma_f32_16x16x32_f16.
//   B operand built on the fly: packed-f16 x tile * splat(g[k,n]) -> pk_mul.
// ---------------------------------------------------------------------------
__global__ __launch_bounds__(256) void gemm_kernel(
    const float* __restrict__ x, const uint32_t* __restrict__ Wh32,
    const float* __restrict__ gtab, const float* __restrict__ bias,
    float* __restrict__ out) {
  __shared__ half2_t xs[XW][XHPAD];        // x tile, f16 pairs: [col][chanPair]
  __shared__ half2_t as_[2][MTILE][APAD];  // W slice f16 pairs, double buffered
  __shared__ half2_t gsh[K_TAPS][NTILE];   // splatted f16 tap weights

  int tid  = threadIdx.x;
  int b    = blockIdx.z;
  int o0   = blockIdx.y * MTILE;
  int n0   = blockIdx.x * NTILE;
  int wave = tid >> 5, lane = tid & 31;
  int wm = wave >> 2, wn = wave & 3;       // 2 x 4 wave grid
  int h = lane >> 4, ln = lane & 15;

  for (int idx = tid; idx < K_TAPS * NTILE; idx += 256) {
    int k = idx / NTILE, nl = idx % NTILE;
    float gv = gtab[((size_t)b * K_TAPS + k) * NPTS + n0 + nl];
    half2_t g2; g2[0] = (_Float16)gv; g2[1] = (_Float16)gv;
    gsh[k][nl] = g2;
  }

  f32x8 acc[4][2];
#pragma unroll
  for (int mi = 0; mi < 4; ++mi)
#pragma unroll
    for (int ni = 0; ni < 2; ++ni)
#pragma unroll
      for (int e = 0; e < 8; ++e) acc[mi][ni][e] = 0.f;

  const float* xb = x + (size_t)b * CIN * NPTS;

  for (int c0 = 0; c0 < CIN; c0 += KSLC) {
    __syncthreads();                       // prior compute done: xs/buf0 reusable
    // stage x tile (f32 -> packed f16), transposed: [col][chanPair]
    for (int idx = tid; idx < (KSLC / 2) * XW; idx += 256) {
      int cp = idx / XW, j = idx % XW;
      int gn = n0 + j - PADH;
      float v0 = 0.f, v1 = 0.f;
      if (gn >= 0 && gn < NPTS) {
        const float* pc = xb + (size_t)(c0 + 2 * cp) * NPTS + gn;
        v0 = pc[0];
        v1 = pc[NPTS];
        if (c0 + KSLC < CIN && (j & 15) == 0) {       // prefetch next K-slice
          __builtin_prefetch(pc + (size_t)KSLC * NPTS, 0, 1);
          __builtin_prefetch(pc + (size_t)(KSLC + 1) * NPTS, 0, 1);
        }
      }
      half2_t hv; hv[0] = (_Float16)v0; hv[1] = (_Float16)v1;
      xs[j][cp] = hv;
    }
    stage_A(as_[0], Wh32, 0, o0, c0, tid);            // prologue: tap 0

    for (int k = 0; k < K_TAPS; ++k) {
      ASYNC_WAIT();
      __syncthreads();                     // x + A[k] staged; buf[(k+1)&1] free
      if (k + 1 < K_TAPS) stage_A(as_[(k + 1) & 1], Wh32, k + 1, o0, c0, tid);
      const half2_t (*ak)[APAD] = as_[k & 1];

      // B fragments (32x16 f16): lane h=0 holds K=2r..2r+1, h=1 holds K=16+2r..
      union BU { half2_t h2[8]; v16h v; } bu[2];
#pragma unroll
      for (int ni = 0; ni < 2; ++ni) {
        int nl2 = wn * 32 + ni * 16 + ln;
        half2_t g2 = gsh[k][nl2];
        int xcol = nl2 + k * DILATION;
#pragma unroll
        for (int r = 0; r < 8; ++r)
          bu[ni].h2[r] = xs[xcol][h * 8 + r] * g2;     // v_pk_mul_f16
      }
#pragma unroll
      for (int mi = 0; mi < 4; ++mi) {
        int m = wm * 64 + mi * 16 + ln;
        union AU { uint32_t u[8]; v16h v; } af;
#pragma unroll
        for (int r2 = 0; r2 < 4; ++r2) {
          // A 16x32 layout: h=0 -> K {0..15}, h=1 -> K {16..31} (matches B)
          int off = 8 * h + 2 * r2;
          uint2 w2 = *(const uint2*)&ak[m][off];
          af.u[2 * r2]     = w2.x;
          af.u[2 * r2 + 1] = w2.y;
        }
#pragma unroll
        for (int ni = 0; ni < 2; ++ni) {
          acc[mi][ni] = __builtin_amdgcn_wmma_f32_16x16x32_f16(
              false, af.v, false, bu[ni].v, (short)0, acc[mi][ni], false, false);
        }
      }
    }
  }

  // Epilogue: + bias, store pre-BN activations
#pragma unroll
  for (int mi = 0; mi < 4; ++mi)
#pragma unroll
    for (int ni = 0; ni < 2; ++ni)
#pragma unroll
      for (int r = 0; r < 8; ++r) {
        int M  = o0 + wm * 64 + mi * 16 + 8 * h + r;
        int Nn = n0 + wn * 32 + ni * 16 + ln;
        out[((size_t)b * COUT + M) * NPTS + Nn] = acc[mi][ni][r] + bias[M];
      }
}

// ---------------------------------------------------------------------------
// Kernel 4: deterministic per-channel batch stats (1 WG per channel)
// ---------------------------------------------------------------------------
__global__ __launch_bounds__(256) void bn_stats_kernel(
    const float* __restrict__ out, float* __restrict__ mean,
    float* __restrict__ invstd) {
  int o = blockIdx.x, tid = threadIdx.x;
  float s = 0.f, ss = 0.f;
  for (int b = 0; b < B_SZ; ++b) {
    const float* p = out + ((size_t)b * COUT + o) * NPTS;
    for (int n = tid; n < NPTS; n += 256) {
      float v = p[n];
      s += v; ss += v * v;
    }
  }
  __shared__ float sh_s[256], sh_q[256];
  sh_s[tid] = s; sh_q[tid] = ss;
  __syncthreads();
  for (int st = 128; st > 0; st >>= 1) {
    if (tid < st) { sh_s[tid] += sh_s[tid + st]; sh_q[tid] += sh_q[tid + st]; }
    __syncthreads();
  }
  if (tid == 0) {
    float inv = 1.0f / (float)(B_SZ * NPTS);
    float m = sh_s[0] * inv;
    float var = sh_q[0] * inv - m * m;
    if (var < 0.f) var = 0.f;
    mean[o]   = m;
    invstd[o] = rsqrtf(var + BN_EPS);
  }
}

// ---------------------------------------------------------------------------
// Kernel 5: normalize + affine + ReLU, in place
// ---------------------------------------------------------------------------
__global__ __launch_bounds__(256) void bn_relu_kernel(
    float* __restrict__ out, const float* __restrict__ mean,
    const float* __restrict__ invstd, const float* __restrict__ gamma,
    const float* __restrict__ beta) {
  size_t idx = (size_t)blockIdx.x * 256 + threadIdx.x;   // over B*COUT*NPTS
  int o = (int)((idx / NPTS) % COUT);
  float v = out[idx];
  v = (v - mean[o]) * invstd[o] * gamma[o] + beta[o];
  out[idx] = v > 0.f ? v : 0.f;
}

// ---------------------------------------------------------------------------
extern "C" void kernel_launch(void* const* d_in, const int* in_sizes, int n_in,
                              void* d_out, int out_size, void* d_ws, size_t ws_size,
                              hipStream_t stream) {
  const float* x      = (const float*)d_in[0];
  const float* coords = (const float*)d_in[1];
  const float* rot    = (const float*)d_in[2];
  const float* dist   = (const float*)d_in[3];
  const float* W      = (const float*)d_in[4];
  const float* bias   = (const float*)d_in[5];
  const float* gamma  = (const float*)d_in[6];
  const float* beta   = (const float*)d_in[7];
  float* out = (float*)d_out;

  char* ws = (char*)d_ws;
  float*    g_tab  = (float*)ws;                               // 2,359,296 B
  uint16_t* Wh     = (uint16_t*)(ws + 2359296);                // 1,179,648 B
  float*    mean   = (float*)(ws + 2359296 + 1179648);         // 1 KB
  float*    invstd = mean + COUT;                              // 1 KB

  tap_weight_kernel<<<(B_SZ * NPTS) / 256, 256, 0, stream>>>(coords, rot, dist, g_tab);
  pack_w_kernel<<<(K_TAPS * COUT * CIN + 255) / 256, 256, 0, stream>>>(W, Wh);
  dim3 gg(NPTS / NTILE, COUT / MTILE, B_SZ);
  gemm_kernel<<<gg, 256, 0, stream>>>(x, (const uint32_t*)Wh, g_tab, bias, out);
  bn_stats_kernel<<<COUT, 256, 0, stream>>>(out, mean, invstd);
  bn_relu_kernel<<<(B_SZ * COUT * NPTS) / 256, 256, 0, stream>>>(out, mean, invstd, gamma, beta);
}
